// SpearmanLoss_79594333929899
// MI455X (gfx1250) — compile-verified, matrix operand order checked by reading, not run on hardware
//
#include <hip/hip_runtime.h>
#include <math.h>

typedef __attribute__((ext_vector_type(16))) _Float16 v16h;
typedef __attribute__((ext_vector_type(8)))  float    v8f;

#define NELEM 16384
#define KF (10.0f * 1.44269504088896340736f)   /* SCALE * log2(e) */

// rank_i = sum_j sigmoid((x_j - x_i)*10) + 1
// One wave32 computes 16 rows; WMMA f32_16x16x32_f16 accumulates row sums
// (B = ones). x kept fully resident in LDS (64KB of 320KB WGP LDS).
__global__ __launch_bounds__(256) void soft_rank_kernel(
    const float* __restrict__ pred,
    const float* __restrict__ target,
    float* __restrict__ ranks /* [2*NELEM] */) {
  __shared__ float xs[NELEM];

  const float* x = (blockIdx.y == 0) ? pred : target;
  float* out = ranks + (size_t)blockIdx.y * NELEM;

  for (int idx = threadIdx.x; idx < NELEM; idx += 256) xs[idx] = x[idx];
  __syncthreads();

  const int lane  = threadIdx.x & 31;
  const int wave  = threadIdx.x >> 5;
  const int iBase = blockIdx.x * 128 + wave * 16;

  // A k order used here (contiguous; any order works since B = ones): lane L -> row M = L%16, K offset (L/16)*16
  const float xi  = xs[iBase + (lane & 15)];
  const float xiK = xi * KF;
  const int  kofs = (lane >> 4) * 16;

  v8f c = {};
  v16h bones;
#pragma unroll
  for (int t = 0; t < 16; ++t) bones[t] = (_Float16)1.0f;

  const float4* xsv = (const float4*)xs;

  for (int j0 = 0; j0 < NELEM; j0 += 32) {
    const int base = (j0 + kofs) >> 2;          // 64B-aligned float4 index
    float4 q0 = xsv[base + 0];
    float4 q1 = xsv[base + 1];
    float4 q2 = xsv[base + 2];
    float4 q3 = xsv[base + 3];
    float xj[16] = {q0.x, q0.y, q0.z, q0.w,
                    q1.x, q1.y, q1.z, q1.w,
                    q2.x, q2.y, q2.z, q2.w,
                    q3.x, q3.y, q3.z, q3.w};
    v16h a;
#pragma unroll
    for (int t = 0; t < 16; ++t) {
      // sigmoid((xj-xi)*10) = 1 / (1 + exp2((xi-xj)*10*log2e))
      float z = xiK - xj[t] * KF;
      float e = __builtin_amdgcn_exp2f(z);
      float s = __builtin_amdgcn_rcpf(1.0f + e);
      a[t] = (_Float16)s;
    }
    // D = A x ones + C  ->  every column of D holds the row sum
    c = __builtin_amdgcn_wmma_f32_16x16x32_f16(
        /*neg_a=*/false, a, /*neg_b=*/false, bones,
        /*c_mod=*/(short)0, c, /*reuse_a=*/false, /*reuse_b=*/false);
  }

  // C/D layout: VGPR r, lanes 0-15 -> M=r ; lanes 16-31 -> M=r+8
  if (lane == 0) {
#pragma unroll
    for (int r = 0; r < 8; ++r) out[iBase + r] = c[r] + 1.0f;
  } else if (lane == 16) {
#pragma unroll
    for (int r = 0; r < 8; ++r) out[iBase + 8 + r] = c[r] + 1.0f;
  }
}

// Single-block finalize: raw moments in f64, loss = 1 - corr
__global__ __launch_bounds__(256) void corr_kernel(
    const float* __restrict__ ranks, float* __restrict__ out) {
  __shared__ double r0[256], r1[256], r2[256], r3[256], r4[256];
  const int tid = threadIdx.x;

  double sp = 0.0, st = 0.0, spp = 0.0, stt = 0.0, spt = 0.0;
  for (int i = tid; i < NELEM; i += 256) {
    double p = (double)ranks[i];
    double t = (double)ranks[NELEM + i];
    sp += p; st += t; spp += p * p; stt += t * t; spt += p * t;
  }
  r0[tid] = sp; r1[tid] = st; r2[tid] = spp; r3[tid] = stt; r4[tid] = spt;
  __syncthreads();
  for (int off = 128; off > 0; off >>= 1) {
    if (tid < off) {
      r0[tid] += r0[tid + off];
      r1[tid] += r1[tid + off];
      r2[tid] += r2[tid + off];
      r3[tid] += r3[tid + off];
      r4[tid] += r4[tid + off];
    }
    __syncthreads();
  }
  if (tid == 0) {
    const double n = (double)NELEM;
    double mp  = r0[0] / n, mt = r1[0] / n;
    double vp  = r2[0] / n - mp * mp;
    double vt  = r3[0] / n - mt * mt;
    double cov = r4[0] / n - mp * mt;
    double corr = cov / (sqrt(vp + 1e-8) * sqrt(vt + 1e-8));
    out[0] = (float)(1.0 - corr);
  }
}

extern "C" void kernel_launch(void* const* d_in, const int* in_sizes, int n_in,
                              void* d_out, int out_size, void* d_ws, size_t ws_size,
                              hipStream_t stream) {
  const float* pred   = (const float*)d_in[0];
  const float* target = (const float*)d_in[1];
  float* ranks = (float*)d_ws;               // 2*NELEM floats = 128 KB scratch

  dim3 grid(NELEM / 128, 2);                 // 128 rows per block; y: pred/target
  soft_rank_kernel<<<grid, 256, 0, stream>>>(pred, target, ranks);
  corr_kernel<<<1, 256, 0, stream>>>(ranks, (float*)d_out);
}
